// Agent_68839735820410
// MI455X (gfx1250) — matrix operand order repaired and checked by reading, not compiled, not run on hardware
//
#include <hip/hip_runtime.h>
#include <math.h>

// ---------------------------------------------------------------------------
// DeepSets forward: phi (per-element MLP, WMMA f16->f32) + sum-pool +
// rho/q heads (VALU) + softmax. One workgroup (256 thr = 8 wave32) per batch.
// Packed B-fragments are staged per-WG into LDS via async-DMA (gfx1250).
// ---------------------------------------------------------------------------

typedef __attribute__((ext_vector_type(16))) _Float16 v16h;
typedef __attribute__((ext_vector_type(8)))  _Float16 v8h;
typedef __attribute__((ext_vector_type(8)))  float    v8f;

typedef v16h __attribute__((address_space(3))) as3_v16h;

#define WMMA_F16(A, B, C) \
  __builtin_amdgcn_wmma_f32_16x16x32_f16(false, (A), false, (B), (short)0, (C), false, false)

#define MB_    128   // elements per batch
#define L1_    120   // phi layer1 out (K for layer2, padded to 128)
#define L2_    100   // phi layer2 out (K for layer3, padded to 128 w/ zeros)
#define L3_    80    // phi layer3 out
#define NT2_   7     // output tiles layer2 (7*16 = 112 >= 100)
#define NT3_   5     // output tiles layer3 (5*16 = 80)
#define KT_    4     // k tiles (4*32 = 128)
#define FRAG_HALFS 512                   // 32 lanes * 16 halfs
#define FRAGS2 (NT2_ * KT_)              // 28
#define FRAGS3 (NT3_ * KT_)              // 20
#define WS_HALFS ((FRAGS2 + FRAGS3) * FRAG_HALFS)   // 24576 halfs = 49152 B
#define WS_BYTES (WS_HALFS * 2)

// ---------------------------------------------------------------------------
// Prep kernel: pre-swizzle f32 weights into per-lane f16 B-fragments in d_ws.
// B order used here (32x16, contiguous k, same as A): lane&15 = column N; lanes 0-15 hold K=0..15 of the
// 32-K tile, lanes 16-31 hold K=16..31; 16 contiguous halfs per lane.
// ---------------------------------------------------------------------------
__global__ __launch_bounds__(256) void Agent_68839735820410_pack(
    const float* __restrict__ w2, const float* __restrict__ w3,
    _Float16* __restrict__ ws) {
  int idx = blockIdx.x * 256 + threadIdx.x;
  if (idx >= WS_HALFS) return;
  int OUT, IN, e;
  const float* W;
  if (idx < FRAGS2 * FRAG_HALFS) { OUT = L2_; IN = L1_; W = w2; e = idx; }
  else { OUT = L3_; IN = L2_; W = w3; e = idx - FRAGS2 * FRAG_HALFS; }
  const int frag = e >> 9, within = e & 511;
  const int lane = within >> 4, i = within & 15;
  const int nT = frag >> 2, kT = frag & 3;
  const int n = nT * 16 + (lane & 15);
  const int k = kT * 32 + ((lane >> 4) * 16) + i;
  float v = (n < OUT && k < IN) ? W[n * IN + k] : 0.f;
  ws[idx] = (_Float16)v;
}

// Fallback: gather+convert a B fragment straight from f32 global weights.
__device__ __forceinline__ v16h load_bfrag_global(const float* __restrict__ W,
                                                  int OUT, int IN, int nT,
                                                  int kT, int lane) {
  const int n  = nT * 16 + (lane & 15);
  const int k0 = kT * 32 + ((lane >> 4) * 16);
  v16h b;
#pragma unroll
  for (int i = 0; i < 16; ++i) {
    const int k = k0 + i;
    float v = 0.f;
    if (n < OUT && k < IN) v = W[n * IN + k];
    b[i] = (_Float16)v;
  }
  return b;
}

// ---------------------------------------------------------------------------
// Main fused kernel. grid = B, block = 256.
// Dynamic LDS (when PACKED): 49152 B fragment buffer.
// ---------------------------------------------------------------------------
template <bool PACKED>
__global__ __launch_bounds__(256) void Agent_68839735820410_main(
    const float* __restrict__ dyn, const float* __restrict__ statc,
    const float* __restrict__ phi_w1, const float* __restrict__ phi_b1,
    const float* __restrict__ phi_w2, const float* __restrict__ phi_b2,
    const float* __restrict__ phi_w3, const float* __restrict__ phi_b3,
    const float* __restrict__ rho_w1, const float* __restrict__ rho_b1,
    const float* __restrict__ rho_w2, const float* __restrict__ rho_b2,
    const float* __restrict__ rho_w3, const float* __restrict__ rho_b3,
    const float* __restrict__ q_w1,  const float* __restrict__ q_b1,
    const float* __restrict__ q_w2,  const float* __restrict__ q_b2,
    const float* __restrict__ q_w3,  const float* __restrict__ q_b3,
    const _Float16* __restrict__ bfr, float* __restrict__ out) {
  __shared__ _Float16 h2s[MB_ * 128];     // 32 KB, f16 activations layer2
  __shared__ float wl1[L1_ * 3];          // W1 staged (K=3, VALU layer)
  __shared__ float bl1[L1_];
  __shared__ float partial[16 * L3_];     // deterministic pool partials
  __shared__ float pooled[L3_];
  __shared__ float rA[200];
  __shared__ float rB[200];
  __shared__ float qv[3];
  extern __shared__ char dynsm[];         // PACKED: 49152 B fragment buffer

  const int tid  = threadIdx.x;
  const int b    = blockIdx.x;
  const int lane = tid & 31;
  const int wave = tid >> 5;

  // ---- stage packed B-fragments into LDS via async DMA (gfx1250 path) -----
  unsigned bflOff = 0;
  if (PACKED) {
    bflOff = (unsigned)(uintptr_t)dynsm;  // low 32 bits == LDS byte offset
    for (int off = tid * 16; off < WS_BYTES; off += 256 * 16) {
      unsigned ldsa = bflOff + (unsigned)off;
      unsigned voff = (unsigned)off;
      asm volatile("global_load_async_to_lds_b128 %0, %1, %2"
                   :: "v"(ldsa), "v"(voff), "s"(bfr)
                   : "memory");
    }
    asm volatile("s_wait_asynccnt 0" ::: "memory");
  }

  for (int i = tid; i < L1_ * 3; i += 256) wl1[i] = phi_w1[i];
  for (int i = tid; i < L1_;     i += 256) bl1[i] = phi_b1[i];
  __syncthreads();

  // ---- per-wave tile: 16 element-rows -------------------------------------
  const int rowBase = wave * 16;
  const int r     = lane & 15;            // A-matrix row within tile
  const int kBase = (lane >> 4) * 8;      // A k offset is 2*kBase = 16h (contiguous)
  const int grow  = rowBase + r;

  const float x0 = dyn[(b * MB_ + grow) * 3 + 0];
  const float x1 = dyn[(b * MB_ + grow) * 3 + 1];
  const float x2 = dyn[(b * MB_ + grow) * 3 + 2];

  // Layer 1 (K=3) computed straight into WMMA A-fragment registers.
  v16h a2[KT_];
#pragma unroll
  for (int kT = 0; kT < KT_; ++kT) {
#pragma unroll
    for (int i = 0; i < 16; ++i) {
      const int k = kT * 32 + 2 * kBase + i;
      float v = 0.f;
      if (k < L1_) {
        v = bl1[k] + wl1[k * 3 + 0] * x0 + wl1[k * 3 + 1] * x1 +
            wl1[k * 3 + 2] * x2;
        v = fmaxf(v, 0.f);
      }
      a2[kT][i] = (_Float16)v;
    }
  }

  const int col  = lane & 15;             // C/D column
  const int mOff = (lane >> 4) * 8;       // C/D row group

  // LDS-resident fragment tables (PACKED path)
  const as3_v16h* bl2 =
      (const as3_v16h*)(uintptr_t)bflOff;
  const as3_v16h* bl3 =
      (const as3_v16h*)(uintptr_t)(bflOff + FRAGS2 * FRAG_HALFS * 2);

  // Preload biases (hide latency away from accumulator drain)
  float bias2[NT2_], bias3[NT3_];
#pragma unroll
  for (int nT = 0; nT < NT2_; ++nT) {
    const int cg = nT * 16 + col;
    bias2[nT] = (cg < L2_) ? phi_b2[cg] : 0.f;
  }
#pragma unroll
  for (int nT = 0; nT < NT3_; ++nT) bias3[nT] = phi_b3[nT * 16 + col];

  // zero pad cols 112..127 of this wave's h2 rows (never written below)
  for (int j = lane; j < 16 * 16; j += 32)
    h2s[(rowBase + (j >> 4)) * 128 + 112 + (j & 15)] = (_Float16)0.f;

  // ---- layer 2: [16x120] x [120x100] via 7x4 WMMAs, pipelined B loads -----
  {
    v16h bf[KT_];
#pragma unroll
    for (int kT = 0; kT < KT_; ++kT)
      bf[kT] = PACKED ? (v16h)bl2[kT * 32 + lane]
                      : load_bfrag_global(phi_w2, L2_, L1_, 0, kT, lane);
#pragma unroll
    for (int nT = 0; nT < NT2_; ++nT) {
      v16h nxt[KT_];
      if (nT + 1 < NT2_) {
#pragma unroll
        for (int kT = 0; kT < KT_; ++kT)
          nxt[kT] = PACKED
                        ? (v16h)bl2[((nT + 1) * KT_ + kT) * 32 + lane]
                        : load_bfrag_global(phi_w2, L2_, L1_, nT + 1, kT, lane);
      }
      v8f acc = {};
#pragma unroll
      for (int kT = 0; kT < KT_; ++kT) acc = WMMA_F16(a2[kT], bf[kT], acc);
      const int cg = nT * 16 + col;
#pragma unroll
      for (int rr = 0; rr < 8; ++rr) {
        const float v = (cg < L2_) ? fmaxf(acc[rr] + bias2[nT], 0.f) : 0.f;
        h2s[(rowBase + mOff + rr) * 128 + cg] = (_Float16)v;
      }
#pragma unroll
      for (int kT = 0; kT < KT_; ++kT) bf[kT] = nxt[kT];
    }
  }

  // ---- layer 3 A-fragments from LDS (two ds_load_b128 per fragment) -------
  v16h a3[KT_];
#pragma unroll
  for (int kT = 0; kT < KT_; ++kT) {
    const v8h lo = *(const v8h*)&h2s[grow * 128 + kT * 32 + 2 * kBase];
    const v8h hi = *(const v8h*)&h2s[grow * 128 + kT * 32 + 8 + 2 * kBase];
#pragma unroll
    for (int i = 0; i < 8; ++i) { a3[kT][i] = lo[i]; a3[kT][8 + i] = hi[i]; }
  }

  // ---- layer 3 + in-register row pooling, pipelined B loads ---------------
  {
    v16h bf[KT_];
#pragma unroll
    for (int kT = 0; kT < KT_; ++kT)
      bf[kT] = PACKED ? (v16h)bl3[kT * 32 + lane]
                      : load_bfrag_global(phi_w3, L3_, L2_, 0, kT, lane);
#pragma unroll
    for (int nT = 0; nT < NT3_; ++nT) {
      v16h nxt[KT_];
      if (nT + 1 < NT3_) {
#pragma unroll
        for (int kT = 0; kT < KT_; ++kT)
          nxt[kT] = PACKED
                        ? (v16h)bl3[((nT + 1) * KT_ + kT) * 32 + lane]
                        : load_bfrag_global(phi_w3, L3_, L2_, nT + 1, kT, lane);
      }
      v8f acc = {};
#pragma unroll
      for (int kT = 0; kT < KT_; ++kT) acc = WMMA_F16(a3[kT], bf[kT], acc);
      const int cg = nT * 16 + col;          // always < 80
      float s = 0.f;
#pragma unroll
      for (int rr = 0; rr < 8; ++rr) s += fmaxf(acc[rr] + bias3[nT], 0.f);
      partial[(wave * 2 + (lane >> 4)) * L3_ + cg] = s;   // 8-row partial sum
#pragma unroll
      for (int kT = 0; kT < KT_; ++kT) bf[kT] = nxt[kT];
    }
  }
  __syncthreads();

  // ---- deterministic pool over 16 partials --------------------------------
  if (tid < L3_) {
    float s = 0.f;
#pragma unroll
    for (int i = 0; i < 16; ++i) s += partial[i * L3_ + tid];
    pooled[tid] = s;
  }
  __syncthreads();

  // ---- rho head: 80 -> 60 -> 60 -> 40 -------------------------------------
  if (tid < 60) {
    float s = rho_b1[tid];
    for (int k = 0; k < 80; ++k) s += rho_w1[tid * 80 + k] * pooled[k];
    rA[tid] = fmaxf(s, 0.f);
  }
  __syncthreads();
  if (tid < 60) {
    float s = rho_b2[tid];
    for (int k = 0; k < 60; ++k) s += rho_w2[tid * 60 + k] * rA[k];
    rB[tid] = fmaxf(s, 0.f);
  }
  __syncthreads();
  if (tid < 40) {
    float s = rho_b3[tid];
    for (int k = 0; k < 60; ++k) s += rho_w3[tid * 60 + k] * rB[k];
    rA[tid] = s;                            // no activation
  } else if (tid < 43) {
    rA[tid] = statc[b * 3 + (tid - 40)];    // concat static
  }
  __syncthreads();

  // ---- q head: 43 -> 200 -> 100 -> 3, softmax -----------------------------
  if (tid < 200) {
    float s = q_b1[tid];
    for (int k = 0; k < 43; ++k) s += q_w1[tid * 43 + k] * rA[k];
    rB[tid] = fmaxf(s, 0.f);
  }
  __syncthreads();
  if (tid < 100) {
    float s = q_b2[tid];
    for (int k = 0; k < 200; ++k) s += q_w2[tid * 200 + k] * rB[k];
    rA[tid] = fmaxf(s, 0.f);
  }
  __syncthreads();
  if (tid < 3) {
    float s = q_b3[tid];
    for (int k = 0; k < 100; ++k) s += q_w3[tid * 100 + k] * rA[k];
    qv[tid] = s;
  }
  __syncthreads();
  if (tid < 3) {
    const float m  = fmaxf(fmaxf(qv[0], qv[1]), qv[2]);
    const float e0 = expf(qv[0] - m), e1 = expf(qv[1] - m), e2 = expf(qv[2] - m);
    const float inv = 1.f / (e0 + e1 + e2);
    const float e = (tid == 0) ? e0 : (tid == 1 ? e1 : e2);
    out[b * 3 + tid] = e * inv;
  }
}

// ---------------------------------------------------------------------------
extern "C" void kernel_launch(void* const* d_in, const int* in_sizes, int n_in,
                              void* d_out, int out_size, void* d_ws,
                              size_t ws_size, hipStream_t stream) {
  const float* dyn    = (const float*)d_in[0];
  const float* statc  = (const float*)d_in[1];
  const float* phi_w1 = (const float*)d_in[2];
  const float* phi_b1 = (const float*)d_in[3];
  const float* phi_w2 = (const float*)d_in[4];
  const float* phi_b2 = (const float*)d_in[5];
  const float* phi_w3 = (const float*)d_in[6];
  const float* phi_b3 = (const float*)d_in[7];
  const float* rho_w1 = (const float*)d_in[8];
  const float* rho_b1 = (const float*)d_in[9];
  const float* rho_w2 = (const float*)d_in[10];
  const float* rho_b2 = (const float*)d_in[11];
  const float* rho_w3 = (const float*)d_in[12];
  const float* rho_b3 = (const float*)d_in[13];
  const float* q_w1   = (const float*)d_in[14];
  const float* q_b1   = (const float*)d_in[15];
  const float* q_w2   = (const float*)d_in[16];
  const float* q_b2   = (const float*)d_in[17];
  const float* q_w3   = (const float*)d_in[18];
  const float* q_b3   = (const float*)d_in[19];
  float* out = (float*)d_out;

  const int B = in_sizes[1] / 3;  // static is [B,3]

  if (ws_size >= (size_t)WS_BYTES) {
    _Float16* ws16 = (_Float16*)d_ws;
    Agent_68839735820410_pack<<<(WS_HALFS + 255) / 256, 256, 0, stream>>>(
        phi_w2, phi_w3, ws16);
    Agent_68839735820410_main<true><<<B, 256, WS_BYTES, stream>>>(
        dyn, statc, phi_w1, phi_b1, phi_w2, phi_b2, phi_w3, phi_b3,
        rho_w1, rho_b1, rho_w2, rho_b2, rho_w3, rho_b3,
        q_w1, q_b1, q_w2, q_b2, q_w3, q_b3, (const _Float16*)ws16, out);
  } else {
    Agent_68839735820410_main<false><<<B, 256, 0, stream>>>(
        dyn, statc, phi_w1, phi_b1, phi_w2, phi_b2, phi_w3, phi_b3,
        rho_w1, rho_b1, rho_w2, rho_b2, rho_w3, rho_b3,
        q_w1, q_b1, q_w2, q_b2, q_w3, q_b3, (const _Float16*)nullptr, out);
  }
}
